// LSTMLayer_63110249447911
// MI455X (gfx1250) — matrix operand order repaired and checked by reading, not compiled, not run on hardware
//
#include <hip/hip_runtime.h>
#include <hip/hip_bf16.h>
#include <math.h>

// ---------------------------------------------------------------------------
// LSTM layer for MI455X (gfx1250), wave32 + WMMA bf16 16x16x32.
//   B=32, T=512, D=1024, H=1024 (4H=4096)
// Phases:
//   1) pack W/U/Wo -> bf16 WMMA B-fragment layout; pack x -> bf16 A-fragments
//   2) x_proj = x@W + b  (big GEMM, 2x2 register-blocked tiles per wave:
//      4 WMMAs per 4 fragment loads = 16 FLOP/byte), stored as f32
//      C-fragments so phase 3 can bias-init its accumulator directly.
//   3) persistent recurrence kernel, 32 blocks x 512 thr, device-wide
//      barrier per timestep; the block's 256 KB U slice is staged ONCE in
//      LDS (320 KB/WGP on CDNA5) and served by ds_load_b128 on the critical
//      path; h is kept ONLY as bf16 A-fragments inside the `states` buffer
//      (mtile slots double as the h double-buffer).
//   4) out = relu(states@Wo + bo), 2x2-blocked WMMA GEMM, scatter to [B,T,H].
// ---------------------------------------------------------------------------

typedef __bf16 bf16_t;
typedef bf16_t v16bf __attribute__((ext_vector_type(16)));
typedef float  v8f   __attribute__((ext_vector_type(8)));

#define Bn   32
#define Tn   512
#define Dn   1024
#define Hn   1024
#define H4n  4096

// workspace layout (bytes), all 256B aligned
#define WPACK_OFF   ((size_t)0)                    // D x 4H bf16  = 8 MB
#define UPACK_OFF   ((size_t)8388608)              // H x 4H bf16  = 8 MB
#define WOPACK_OFF  ((size_t)16777216)             // H x H  bf16  = 2 MB
#define BIAS4_OFF   ((size_t)18874368)             // 4H f32       = 16 KB
#define CBUF_OFF    ((size_t)18890752)             // 32x1024 f32  = 128 KB
#define CNT_OFF     ((size_t)19021824)             // barrier ctr  (256 B)
#define SA_OFF      ((size_t)19022080)             // (2+1024) mtiles A-frag bf16
#define XPROJ_OFF   ((size_t)52642048)             // T*B*4H f32 C-frags = 256 MB
#define XPACK_OFF   ((size_t)321077504)            // T*B*D bf16 A-frags = 32 MB

#define SA_MTILE_ELTS  16384                       // 32 ksteps * 32 lanes * 16
#define FRAG_STRIDE    512                         // 32 lanes * 16 elts per kstep
#define TILE_ELTS      16384                       // 32 ksteps * FRAG_STRIDE

// recurrence dynamic LDS: 8 B-tiles (256 KB) + z exchange (16 KB)
#define REC_LDS_U_BYTES  (8 * TILE_ELTS * 2)       // 262144
#define REC_LDS_BYTES    (REC_LDS_U_BYTES + 16 * 32 * 8 * 4)  // 278528

__device__ __forceinline__ int kmapA(int lane, int i) {
  // 16-bit A-matrix 16x32 lane/K mapping (ISA 7.12.2)
  return (i & 7) + ((i >> 3) << 4) + ((lane >> 4) << 3);
}

__device__ __forceinline__ v8f wmma_bf16(v16bf a, v16bf b, v8f c) {
  return __builtin_amdgcn_wmma_f32_16x16x32_bf16(false, a, false, b,
                                                 (short)0, c, false, false);
}

// ---- pack a 1024x1024 f32 row-major matrix into bf16 B-fragments ----------
__global__ __launch_bounds__(256) void pack_b_kernel(
    const float* __restrict__ src, bf16_t* __restrict__ dst, int ntileOffset) {
  int tid  = blockIdx.x * 256 + threadIdx.x;       // < 1<<20
  int i    = tid & 15;
  int lane = (tid >> 4) & 31;
  int ks   = (tid >> 9) & 31;
  int ntl  = tid >> 14;                            // 0..63
  int n = ntl * 16 + (lane & 15);
  int k = ks * 32 + kmapA(lane, i);                // B: same K mapping as A (kmapA)
  float v = src[k * 1024 + n];
  dst[((((size_t)(ntileOffset + ntl)) * 32 + ks) * 32 + lane) * 16 + i] = (bf16_t)v;
}

// ---- pack x [B,T,D] into bf16 A-fragments, rows r = t*32 + b --------------
__global__ __launch_bounds__(256) void pack_ax_kernel(
    const float* __restrict__ x, bf16_t* __restrict__ dst) {
  int tid  = blockIdx.x * 256 + threadIdx.x;       // < 1<<24
  int i    = tid & 15;
  int lane = (tid >> 4) & 31;
  int ks   = (tid >> 9) & 31;
  int mt   = tid >> 14;                            // 0..1023
  int row  = mt * 16 + (lane & 15);
  int b    = row & 31;
  int t    = row >> 5;
  int k    = ks * 32 + kmapA(lane, i);
  dst[tid] = (bf16_t)x[((size_t)(b * Tn + t)) * Dn + k];
}

// ---- concatenate the 4 gate biases --------------------------------------
__global__ void bias_concat_kernel(const float* __restrict__ bi,
                                   const float* __restrict__ bf,
                                   const float* __restrict__ bg,
                                   const float* __restrict__ bc,
                                   float* __restrict__ dst) {
  int tid = blockIdx.x * 256 + threadIdx.x;
  if (tid >= H4n) return;
  int g = tid >> 10, n = tid & 1023;
  const float* s = (g == 0) ? bi : (g == 1) ? bf : (g == 2) ? bg : bc;
  dst[tid] = s[n];
}

// ---- phase 2: x_proj = x@W + b, 2x2-blocked, stored as f32 C-fragments ----
__global__ __launch_bounds__(256) void gemm_xproj_kernel(
    const bf16_t* __restrict__ xpack, const bf16_t* __restrict__ wpack,
    const float* __restrict__ bias, float* __restrict__ xproj) {
  int wave = (blockIdx.x * 256 + threadIdx.x) >> 5;  // 0..65535
  int lane = threadIdx.x & 31;
  int mtile0 = (wave >> 7) << 1;                     // 0,2,..,1022
  int ntile0 = (wave & 127) << 1;                    // 0,2,..,254
  float bv0 = bias[ntile0 * 16 + (lane & 15)];
  float bv1 = bias[(ntile0 + 1) * 16 + (lane & 15)];
  v8f acc00, acc01, acc10, acc11;
#pragma unroll
  for (int r = 0; r < 8; ++r) {
    acc00[r] = bv0; acc01[r] = bv1; acc10[r] = bv0; acc11[r] = bv1;
  }
  const bf16_t* ap0 = xpack + (((size_t)mtile0 * 32) * 32 + lane) * 16;
  const bf16_t* ap1 = ap0 + (size_t)TILE_ELTS;
  const bf16_t* bp0 = wpack + (((size_t)ntile0 * 32) * 32 + lane) * 16;
  const bf16_t* bp1 = bp0 + (size_t)TILE_ELTS;
#pragma unroll 2
  for (int ks = 0; ks < 32; ++ks) {
    if (ks + 1 < 32) {
      __builtin_prefetch(bp0 + FRAG_STRIDE, 0, 1);
      __builtin_prefetch(bp1 + FRAG_STRIDE, 0, 1);
    }
    v16bf a0 = *(const v16bf*)ap0;
    v16bf a1 = *(const v16bf*)ap1;
    v16bf b0 = *(const v16bf*)bp0;
    v16bf b1 = *(const v16bf*)bp1;
    acc00 = wmma_bf16(a0, b0, acc00);
    acc01 = wmma_bf16(a0, b1, acc01);
    acc10 = wmma_bf16(a1, b0, acc10);
    acc11 = wmma_bf16(a1, b1, acc11);
    ap0 += FRAG_STRIDE; ap1 += FRAG_STRIDE;
    bp0 += FRAG_STRIDE; bp1 += FRAG_STRIDE;
  }
  // fragment index layout stays (mtile*256 + ntile) for phase 3
  *(v8f*)(xproj + ((size_t)(mtile0 * 256 + ntile0) * 32 + lane) * 8)       = acc00;
  *(v8f*)(xproj + ((size_t)(mtile0 * 256 + ntile0 + 1) * 32 + lane) * 8)   = acc01;
  *(v8f*)(xproj + ((size_t)((mtile0 + 1) * 256 + ntile0) * 32 + lane) * 8) = acc10;
  *(v8f*)(xproj + ((size_t)((mtile0 + 1) * 256 + ntile0 + 1) * 32 + lane) * 8) = acc11;
}

// ---- phase 3: persistent recurrence --------------------------------------
// 32 blocks (one per 32-column slice of H) x 512 threads (16 waves).
// wave w: gate=w>>2, nt=(w>>1)&1, mt=w&1 -> one 16x16 z tile.
// U slice (8 B-tiles, 256 KB) staged once in LDS; B frags via ds_load_b128.
// h lives as bf16 A-fragments inside statesA: h_in(t) at mtiles {2t, 2t+1}
// (mtiles 0,1 are pre-zeroed = h0); step t writes mtiles {2t+2, 2t+3}.
__global__ __launch_bounds__(512) void lstm_rec_kernel(
    const bf16_t* __restrict__ upack, const float* __restrict__ xproj,
    float* __restrict__ cbuf, bf16_t* __restrict__ statesA,
    unsigned* __restrict__ cnt) {
  extern __shared__ char dynsmem[];
  bf16_t* ldsU = (bf16_t*)dynsmem;                   // 8 tiles * 16384 elts
  float*  zsh  = (float*)(dynsmem + REC_LDS_U_BYTES); // 16 KB z-tile exchange
  const int tid  = threadIdx.x;
  const int lane = tid & 31;
  const int w    = tid >> 5;                         // 0..15
  const int gate = w >> 2;
  const int nt   = (w >> 1) & 1;
  const int mt   = w & 1;
  const int lt   = w >> 1;                           // local B tile: gate*2+nt
  const int wg   = blockIdx.x;                       // 0..31: H column slice
  const int ntileU = gate * 64 + wg * 2 + nt;        // B tile index in 4H
  const unsigned nblk = gridDim.x;

  // stage this block's U slice into LDS (4 gates x 2 adjacent tiles, 256 KB)
#pragma unroll
  for (int g = 0; g < 4; ++g) {
    const bf16_t* src = upack + ((size_t)(g * 64 + wg * 2)) * TILE_ELTS;
    bf16_t* dst = ldsU + g * (2 * TILE_ELTS);
    for (int e = tid; e < 2048; e += 512) {          // 2048 chunks of 32 B
      *(v16bf*)(dst + e * 16) = *(const v16bf*)(src + e * 16);
    }
  }
  __syncthreads();

  const bf16_t* bU = ldsU + ((size_t)lt * 32 * 32 + lane) * 16;

  for (int t = 0; t < Tn; ++t) {
    const int mtA = 2 * t + mt;                      // h_in(t) A-frag mtile
    // accumulator = x_proj tile (bias already folded in)
    v8f acc = *(const v8f*)(xproj +
        ((size_t)(mtA * 256 + ntileU) * 32 + lane) * 8);
    const bf16_t* ap = statesA + (((size_t)mtA * 32) * 32 + lane) * 16;
#pragma unroll 4
    for (int ks = 0; ks < 32; ++ks) {
      v16bf a  = *(const v16bf*)ap;
      v16bf bm = *(const v16bf*)(bU + (size_t)ks * FRAG_STRIDE);
      acc = wmma_bf16(a, bm, acc);
      ap += FRAG_STRIDE;
    }
    // exchange z tiles across the 4 gates
    *(v8f*)(zsh + (w * 32 + lane) * 8) = acc;
    __syncthreads();
    if (w < 4) {                                     // gate-0 waves do gating
      const int emt = w & 1, ent = (w >> 1) & 1;
      const int mtileS = 2 * t + 2 + emt;            // h_out mtile slot
#pragma unroll
      for (int r = 0; r < 8; ++r) {
        float zi = zsh[((0 * 4 + w) * 32 + lane) * 8 + r];
        float zf = zsh[((1 * 4 + w) * 32 + lane) * 8 + r];
        float zg = zsh[((2 * 4 + w) * 32 + lane) * 8 + r];
        float zc = zsh[((3 * 4 + w) * 32 + lane) * 8 + r];
        int m  = r + ((lane >> 4) << 3);
        int b  = emt * 16 + m;
        int nn = lane & 15;
        int kk = ent * 16 + nn;                      // K offset in wg's slice
        int j  = wg * 32 + kk;
        float ig = 1.0f / (1.0f + __expf(-zi));
        float fg = 1.0f / (1.0f + __expf(-zf));
        float gg = 1.0f / (1.0f + __expf(-zg));      // ref applies sigmoid to g
        float ct = tanhf(zc);
        int cidx = b * Hn + j;
        float cn = fg * cbuf[cidx] + ig * ct;
        cbuf[cidx] = cn;
        float hn = gg * tanhf(cn);
        // scatter into A-fragment layout (inverse of kmapA)
        int la = m + (((kk >> 3) & 1) << 4);
        int ie = (kk & 7) + (((kk >> 4) & 1) << 3);
        statesA[(((size_t)mtileS * 32 + wg) * 32 + la) * 16 + ie] = (bf16_t)hn;
      }
    }
    // device-wide barrier: make h_out visible to all blocks for step t+1
    __threadfence();
    __syncthreads();
    if (tid == 0) {
      atomicAdd(cnt, 1u);
      unsigned target = (unsigned)(t + 1) * nblk;
      while (*(volatile unsigned*)cnt < target) {
        __builtin_amdgcn_s_sleep(1);
      }
    }
    __syncthreads();
    __threadfence();
  }
}

// ---- phase 4: out = relu(states @ Wo + bo), 2x2-blocked, out [B,T,H] f32 --
__global__ __launch_bounds__(256) void gemm_out_kernel(
    const bf16_t* __restrict__ statesA, const bf16_t* __restrict__ wopack,
    const float* __restrict__ bo, float* __restrict__ out) {
  int wave = (blockIdx.x * 256 + threadIdx.x) >> 5;  // 0..16383
  int lane = threadIdx.x & 31;
  int mtile0 = (wave >> 5) << 1;                     // 0,2,..,1022
  int ntile0 = (wave & 31) << 1;                     // 0,2,..,62
  float bv0 = bo[ntile0 * 16 + (lane & 15)];
  float bv1 = bo[(ntile0 + 1) * 16 + (lane & 15)];
  v8f acc00, acc01, acc10, acc11;
#pragma unroll
  for (int r = 0; r < 8; ++r) {
    acc00[r] = bv0; acc01[r] = bv1; acc10[r] = bv0; acc11[r] = bv1;
  }
  // +2 mtile offset: slots 0,1 are h0; outputs start at slot 2
  const bf16_t* ap0 = statesA + (((size_t)(mtile0 + 2) * 32) * 32 + lane) * 16;
  const bf16_t* ap1 = ap0 + (size_t)TILE_ELTS;
  const bf16_t* bp0 = wopack + (((size_t)ntile0 * 32) * 32 + lane) * 16;
  const bf16_t* bp1 = bp0 + (size_t)TILE_ELTS;
#pragma unroll 2
  for (int ks = 0; ks < 32; ++ks) {
    if (ks + 1 < 32) {
      __builtin_prefetch(bp0 + FRAG_STRIDE, 0, 1);
      __builtin_prefetch(bp1 + FRAG_STRIDE, 0, 1);
    }
    v16bf a0 = *(const v16bf*)ap0;
    v16bf a1 = *(const v16bf*)ap1;
    v16bf b0 = *(const v16bf*)bp0;
    v16bf b1 = *(const v16bf*)bp1;
    acc00 = wmma_bf16(a0, b0, acc00);
    acc01 = wmma_bf16(a0, b1, acc01);
    acc10 = wmma_bf16(a1, b0, acc10);
    acc11 = wmma_bf16(a1, b1, acc11);
    ap0 += FRAG_STRIDE; ap1 += FRAG_STRIDE;
    bp0 += FRAG_STRIDE; bp1 += FRAG_STRIDE;
  }
#pragma unroll
  for (int q = 0; q < 4; ++q) {
    int mtile = mtile0 + (q >> 1);
    int ntile = ntile0 + (q & 1);
    v8f acc = (q == 0) ? acc00 : (q == 1) ? acc01 : (q == 2) ? acc10 : acc11;
    int n = ntile * 16 + (lane & 15);
#pragma unroll
    for (int r = 0; r < 8; ++r) {
      int m   = r + ((lane >> 4) << 3);
      int row = mtile * 16 + m;                      // row = t*32 + b
      int b   = row & 31;
      int t   = row >> 5;
      float v = acc[r];
      out[((size_t)(b * Tn + t)) * Hn + n] = v > 0.0f ? v : 0.0f;
    }
  }
}

extern "C" void kernel_launch(void* const* d_in, const int* in_sizes, int n_in,
                              void* d_out, int out_size, void* d_ws, size_t ws_size,
                              hipStream_t stream) {
  (void)in_sizes; (void)n_in; (void)out_size; (void)ws_size;
  char* ws = (char*)d_ws;
  bf16_t*  wpack   = (bf16_t*)(ws + WPACK_OFF);
  bf16_t*  upack   = (bf16_t*)(ws + UPACK_OFF);
  bf16_t*  wopack  = (bf16_t*)(ws + WOPACK_OFF);
  float*   bias4   = (float*)(ws + BIAS4_OFF);
  float*   cbuf    = (float*)(ws + CBUF_OFF);
  unsigned* cnt    = (unsigned*)(ws + CNT_OFF);
  bf16_t*  statesA = (bf16_t*)(ws + SA_OFF);
  float*   xproj   = (float*)(ws + XPROJ_OFF);
  bf16_t*  xpack   = (bf16_t*)(ws + XPACK_OFF);

  const float* x   = (const float*)d_in[0];
  const float* W_i = (const float*)d_in[1];
  const float* U_i = (const float*)d_in[2];
  const float* W_f = (const float*)d_in[3];
  const float* U_f = (const float*)d_in[4];
  const float* W_g = (const float*)d_in[5];
  const float* U_g = (const float*)d_in[6];
  const float* W_c = (const float*)d_in[7];
  const float* U_c = (const float*)d_in[8];
  const float* W_o = (const float*)d_in[9];
  const float* b_i = (const float*)d_in[10];
  const float* b_f = (const float*)d_in[11];
  const float* b_g = (const float*)d_in[12];
  const float* b_c = (const float*)d_in[13];
  const float* b_o = (const float*)d_in[14];
  float* out = (float*)d_out;

  // zero: c-state (128KB) + barrier counter (256B) + h0 A-frag slots (64KB)
  hipMemsetAsync(ws + CBUF_OFF, 0,
                 (SA_OFF - CBUF_OFF) + 2 * SA_MTILE_ELTS * sizeof(bf16_t),
                 stream);

  // pack weights into WMMA B-fragment layouts (bf16)
  pack_b_kernel<<<4096, 256, 0, stream>>>(W_i, wpack, 0);
  pack_b_kernel<<<4096, 256, 0, stream>>>(W_f, wpack, 64);
  pack_b_kernel<<<4096, 256, 0, stream>>>(W_g, wpack, 128);
  pack_b_kernel<<<4096, 256, 0, stream>>>(W_c, wpack, 192);
  pack_b_kernel<<<4096, 256, 0, stream>>>(U_i, upack, 0);
  pack_b_kernel<<<4096, 256, 0, stream>>>(U_f, upack, 64);
  pack_b_kernel<<<4096, 256, 0, stream>>>(U_g, upack, 128);
  pack_b_kernel<<<4096, 256, 0, stream>>>(U_c, upack, 192);
  pack_b_kernel<<<4096, 256, 0, stream>>>(W_o, wopack, 0);
  bias_concat_kernel<<<16, 256, 0, stream>>>(b_i, b_f, b_g, b_c, bias4);
  pack_ax_kernel<<<65536, 256, 0, stream>>>(x, xpack);

  // phase 2: fused input projection (16384 x 4096 x 1024 GEMM, 2x2 blocked)
  gemm_xproj_kernel<<<8192, 256, 0, stream>>>(xpack, wpack, bias4, xproj);

  // phase 3: persistent recurrence (512 steps, device-wide barrier per step,
  // 272 KB dynamic LDS for the U slice + z exchange)
  lstm_rec_kernel<<<32, 512, REC_LDS_BYTES, stream>>>(upack, xproj, cbuf,
                                                      statesA, cnt);

  // phase 4: output projection + ReLU (2x2 blocked)
  gemm_out_kernel<<<2048, 256, 0, stream>>>(statesA, wopack, b_o, out);
}
